// LLaMABlock_46514495815763
// MI455X (gfx1250) — matrix operand order repaired and checked by reading, not compiled, not run on hardware
//
#include <hip/hip_runtime.h>
#include <hip/hip_bf16.h>
#include <math.h>

// ---------------- types ----------------
typedef __attribute__((ext_vector_type(16))) __bf16 v16bf;
typedef __attribute__((ext_vector_type(8)))  float  v8f;

#define B_    2
#define T_    2048
#define E_    2048
#define H_    16
#define KVH_  8
#define DH_   128
#define HID_  5632
#define M_    (B_ * T_)   // 4096 rows

// ---------------- WMMA helpers (CDNA5 wave32 layouts) ----------------
__device__ __forceinline__ v8f wmma_bf16(v16bf a, v16bf b, v8f c) {
  // D(f32 16x16) = A(bf16 16x32) * B(bf16 32x16) + C
  return __builtin_amdgcn_wmma_f32_16x16x32_bf16(false, a, false, b, (short)0, c, false, false);
}

// A fragment: 16x32 bf16, row = lane&15.
// lanes 0-15 hold K = kk+0..7 (u[0]) and kk+16..23 (u[1]);
// lanes 16-31 hold K = kk+8..15 and kk+24..31  (per ISA 7.12.2).
__device__ __forceinline__ v16bf load_frag_a(const __bf16* base, int stride, int kk, int lane) {
  const __bf16* p = base + (size_t)(lane & 15) * stride + kk + ((lane >> 4) << 3);
  union { v16bf v; uint4 u[2]; } t;
  t.u[0] = *(const uint4*)p;
  t.u[1] = *(const uint4*)(p + 16);
  return t.v;
}

// B fragment from a K-contiguous (i.e. transposed-in-LDS) tile Bt[N][K]:
// col n = lane&15; lanes 0-15 hold K rows 0..7 and 16..23, lanes 16-31 hold 8..15 and 24..31.
__device__ __forceinline__ v16bf load_frag_b(const __bf16* bt, int stride, int lane) {
  const __bf16* p = bt + (size_t)(lane & 15) * stride + ((lane >> 4) << 3);
  union { v16bf v; uint4 u[2]; } t;
  t.u[0] = *(const uint4*)p;
  t.u[1] = *(const uint4*)(p + 16);
  return t.v;
}

// ---------------- RMSNorm (fp32 in -> bf16 out) ----------------
__global__ __launch_bounds__(256) void rmsnorm_kernel(const float* __restrict__ x,
                                                      const float* __restrict__ w,
                                                      __bf16* __restrict__ out, int ncols) {
  int row = blockIdx.x;
  const float* xr = x + (size_t)row * ncols;
  float ss = 0.f;
  for (int c = threadIdx.x; c < ncols; c += 256) { float v = xr[c]; ss += v * v; }
  __shared__ float red[256];
  red[threadIdx.x] = ss;
  __syncthreads();
  for (int s = 128; s > 0; s >>= 1) {
    if (threadIdx.x < s) red[threadIdx.x] += red[threadIdx.x + s];
    __syncthreads();
  }
  float scale = rsqrtf(red[0] / (float)ncols + 1e-6f);
  for (int c = threadIdx.x; c < ncols; c += 256)
    out[(size_t)row * ncols + c] = (__bf16)(xr[c] * scale * w[c]);
}

// ---------------- fp32 -> bf16 conversion ----------------
__global__ __launch_bounds__(256) void conv_kernel(const float* __restrict__ in,
                                                   __bf16* __restrict__ out, size_t n) {
  size_t i = (size_t)blockIdx.x * 256 + threadIdx.x;
  if (i < n) out[i] = (__bf16)in[i];
}

// ---------------- generic bf16 GEMM: C(f32) = A(bf16 MxK) @ B(bf16 KxN) [+ Res] ----------------
// Block: 256 thr = 8 waves in 4(M)x2(N); block tile 128x128, BK=64; wave tile 32x64.
__global__ __launch_bounds__(256) void gemm_kernel(const __bf16* __restrict__ A,
                                                   const __bf16* __restrict__ B,
                                                   float* __restrict__ C,
                                                   const float* __restrict__ Res,
                                                   int M, int N, int K) {
  __shared__ __bf16 As[128][72];   // row-major, padded
  __shared__ __bf16 Bt[128][72];   // B transposed: Bt[n][k]
  int tid = threadIdx.x, wave = tid >> 5, lane = tid & 31;
  int wm = (wave & 3) * 32, wn = (wave >> 2) * 64;
  int m0 = blockIdx.y * 128, n0 = blockIdx.x * 128;
  v8f acc[2][4] = {};
  for (int k0 = 0; k0 < K; k0 += 64) {
#pragma unroll
    for (int it = 0; it < 4; it++) {                  // A tile 128x64
      int idx = tid + it * 256;
      int r = idx >> 3, c = (idx & 7) * 8;
      *(uint4*)&As[r][c] = *(const uint4*)(A + (size_t)(m0 + r) * K + k0 + c);
    }
#pragma unroll
    for (int it = 0; it < 4; it++) {                  // B tile 64x128, transpose on store
      int idx = tid + it * 256;
      int r = idx >> 4, c = (idx & 15) * 8;
      uint4 u = *(const uint4*)(B + (size_t)(k0 + r) * N + n0 + c);
      const __bf16* e = (const __bf16*)&u;
#pragma unroll
      for (int j = 0; j < 8; j++) Bt[c + j][r] = e[j];
    }
    __syncthreads();
#pragma unroll
    for (int kk = 0; kk < 64; kk += 32) {
      v16bf a0 = load_frag_a(&As[wm][0], 72, kk, lane);
      v16bf a1 = load_frag_a(&As[wm + 16][0], 72, kk, lane);
      v16bf b[4];
#pragma unroll
      for (int i = 0; i < 4; i++) b[i] = load_frag_b(&Bt[wn + i * 16][kk], 72, lane);
#pragma unroll
      for (int i = 0; i < 4; i++) {
        acc[0][i] = wmma_bf16(a0, b[i], acc[0][i]);
        acc[1][i] = wmma_bf16(a1, b[i], acc[1][i]);
      }
    }
    __syncthreads();
  }
  int col = lane & 15, rb = (lane >> 4) * 8;
#pragma unroll
  for (int im = 0; im < 2; im++)
#pragma unroll
    for (int in = 0; in < 4; in++) {
      int rbase = m0 + wm + im * 16 + rb;
      int cbase = n0 + wn + in * 16 + col;
#pragma unroll
      for (int r = 0; r < 8; r++) {
        size_t off = (size_t)(rbase + r) * N + cbase;
        float v = acc[im][in][r];
        if (Res) v += Res[off];
        C[off] = v;
      }
    }
}

// ---------------- fused gate+up GEMM with SiLU*mul epilogue -> bf16 ----------------
// Block tile 128(M)x64(N), BK=64; 8 waves 4x2; wave tile 32x32 per matrix.
__global__ __launch_bounds__(256) void gateup_kernel(const __bf16* __restrict__ A,
                                                     const __bf16* __restrict__ Bg,
                                                     const __bf16* __restrict__ Bu,
                                                     __bf16* __restrict__ Out,
                                                     int M, int N, int K) {
  __shared__ __bf16 As[128][72];
  __shared__ __bf16 Btg[64][72];
  __shared__ __bf16 Btu[64][72];
  int tid = threadIdx.x, wave = tid >> 5, lane = tid & 31;
  int wm = (wave & 3) * 32, wn = (wave >> 2) * 32;
  int m0 = blockIdx.y * 128, n0 = blockIdx.x * 64;
  v8f ag[2][2] = {}, au[2][2] = {};
  for (int k0 = 0; k0 < K; k0 += 64) {
#pragma unroll
    for (int it = 0; it < 4; it++) {
      int idx = tid + it * 256;
      int r = idx >> 3, c = (idx & 7) * 8;
      *(uint4*)&As[r][c] = *(const uint4*)(A + (size_t)(m0 + r) * K + k0 + c);
    }
#pragma unroll
    for (int it = 0; it < 2; it++) {                  // 64x64 tiles, transpose
      int idx = tid + it * 256;
      int r = idx >> 3, c = (idx & 7) * 8;
      uint4 u0 = *(const uint4*)(Bg + (size_t)(k0 + r) * N + n0 + c);
      uint4 u1 = *(const uint4*)(Bu + (size_t)(k0 + r) * N + n0 + c);
      const __bf16* e0 = (const __bf16*)&u0;
      const __bf16* e1 = (const __bf16*)&u1;
#pragma unroll
      for (int j = 0; j < 8; j++) { Btg[c + j][r] = e0[j]; Btu[c + j][r] = e1[j]; }
    }
    __syncthreads();
#pragma unroll
    for (int kk = 0; kk < 64; kk += 32) {
      v16bf a0 = load_frag_a(&As[wm][0], 72, kk, lane);
      v16bf a1 = load_frag_a(&As[wm + 16][0], 72, kk, lane);
#pragma unroll
      for (int i = 0; i < 2; i++) {
        v16bf bg = load_frag_b(&Btg[wn + i * 16][kk], 72, lane);
        v16bf bu = load_frag_b(&Btu[wn + i * 16][kk], 72, lane);
        ag[0][i] = wmma_bf16(a0, bg, ag[0][i]);
        ag[1][i] = wmma_bf16(a1, bg, ag[1][i]);
        au[0][i] = wmma_bf16(a0, bu, au[0][i]);
        au[1][i] = wmma_bf16(a1, bu, au[1][i]);
      }
    }
    __syncthreads();
  }
  int col = lane & 15, rb = (lane >> 4) * 8;
#pragma unroll
  for (int im = 0; im < 2; im++)
#pragma unroll
    for (int in = 0; in < 2; in++) {
      int rbase = m0 + wm + im * 16 + rb;
      int cbase = n0 + wn + in * 16 + col;
#pragma unroll
      for (int r = 0; r < 8; r++) {
        float g = ag[im][in][r];
        float u = au[im][in][r];
        float s = g / (1.f + __expf(-g));     // SiLU
        Out[(size_t)(rbase + r) * N + cbase] = (__bf16)(s * u);
      }
    }
}

// ---------------- RoPE (interleaved pairs) + [B,T,h,D] -> [B,h,T,D] bf16 ----------------
__global__ __launch_bounds__(256) void rope_kernel(const float* __restrict__ in,
                                                   __bf16* __restrict__ out, int nh) {
  size_t i = (size_t)blockIdx.x * 256 + threadIdx.x;  // over B*T*nh*(DH/2)
  int j = (int)(i & 63);
  size_t rest = i >> 6;
  int h = (int)(rest % nh); rest /= nh;
  int t = (int)(rest % T_);
  int b = (int)(rest / T_);
  float inv_freq = __expf(-(float)(2 * j) * (9.210340371976184f / 128.f)); // 10000^(-2j/128)
  float ang = (float)t * inv_freq;
  float c = cosf(ang), s = sinf(ang);
  const float* p = in + (((size_t)b * T_ + t) * nh + h) * DH_ + 2 * j;
  float xe = p[0], xo = p[1];
  __bf16* q = out + (((size_t)b * nh + h) * T_ + t) * DH_ + 2 * j;
  q[0] = (__bf16)(xe * c - xo * s);
  q[1] = (__bf16)(xe * s + xo * c);
}

__global__ __launch_bounds__(256) void vtrans_kernel(const float* __restrict__ in,
                                                     __bf16* __restrict__ out, int nh) {
  size_t i = (size_t)blockIdx.x * 256 + threadIdx.x;  // over B*T*nh*DH
  int d = (int)(i & 127);
  size_t rest = i >> 7;
  int h = (int)(rest % nh); rest /= nh;
  int t = (int)(rest % T_);
  int b = (int)(rest / T_);
  out[(((size_t)b * nh + h) * T_ + t) * DH_ + d] =
      (__bf16)in[(((size_t)b * T_ + t) * nh + h) * DH_ + d];
}

// ---------------- causal flash attention (bf16 WMMA, f32 accum) ----------------
// grid: (T/64, H, B); block 128 thr = 4 waves; wave owns 16 q rows.
__global__ __launch_bounds__(128) void attn_kernel(const __bf16* __restrict__ Q,
                                                   const __bf16* __restrict__ K,
                                                   const __bf16* __restrict__ V,
                                                   __bf16* __restrict__ Out) {
  __shared__ __bf16 Qs[64][136];
  __shared__ __bf16 Ks[32][136];
  __shared__ __bf16 Vst[128][40];        // V transposed: Vst[d][key]
  __shared__ float  Ss[4][16][32];       // raw scores per wave
  __shared__ __bf16 Ps[4][16][40];       // exp probs per wave (A-frag layout source)
  __shared__ float  mrow[4][16], lrow[4][16], arow[4][16];

  int tid = threadIdx.x, wave = tid >> 5, lane = tid & 31;
  int b = blockIdx.z, h = blockIdx.y, q0 = blockIdx.x * 64;
  int kvh = h >> 1;  // GQA: H/KVH = 2

  const __bf16* Qp = Q + (((size_t)b * H_ + h) * T_ + q0) * DH_;
  const __bf16* Kp = K + ((size_t)b * KVH_ + kvh) * (size_t)T_ * DH_;
  const __bf16* Vp = V + ((size_t)b * KVH_ + kvh) * (size_t)T_ * DH_;

#pragma unroll
  for (int it = 0; it < 8; it++) {                    // Q tile 64x128
    int idx = tid + it * 128;
    int r = idx >> 4, c = (idx & 15) * 8;
    *(uint4*)&Qs[r][c] = *(const uint4*)(Qp + (size_t)r * DH_ + c);
  }
  if (lane < 16) { mrow[wave][lane] = -1e30f; lrow[wave][lane] = 0.f; }
  __syncthreads();

  v8f O[8] = {};
  const float inv_scale = 0.08838834764831845f;       // 1/sqrt(128)
  int col = lane & 15, rb = (lane >> 4) * 8;
  int kend = q0 + 64;

  for (int kt = 0; kt < kend; kt += 32) {
    __syncthreads();
#pragma unroll
    for (int it = 0; it < 4; it++) {                  // K tile 32x128
      int idx = tid + it * 128;
      int r = idx >> 4, c = (idx & 15) * 8;
      *(uint4*)&Ks[r][c] = *(const uint4*)(Kp + (size_t)(kt + r) * DH_ + c);
    }
#pragma unroll
    for (int it = 0; it < 4; it++) {                  // V tile 32x128 -> transpose
      int idx = tid + it * 128;
      int r = idx >> 4, c = (idx & 15) * 8;
      uint4 u = *(const uint4*)(Vp + (size_t)(kt + r) * DH_ + c);
      const __bf16* e = (const __bf16*)&u;
#pragma unroll
      for (int j = 0; j < 8; j++) Vst[c + j][r] = e[j];
    }
    __syncthreads();

    // S = Q @ K^T (contraction over d=128)
    v8f s0 = {}, s1 = {};
#pragma unroll
    for (int kk = 0; kk < DH_; kk += 32) {
      v16bf aq = load_frag_a(&Qs[wave * 16][0], 136, kk, lane);
      v16bf b0 = load_frag_b(&Ks[0][kk], 136, lane);   // keys 0..15
      v16bf b1 = load_frag_b(&Ks[16][kk], 136, lane);  // keys 16..31
      s0 = wmma_bf16(aq, b0, s0);
      s1 = wmma_bf16(aq, b1, s1);
    }
#pragma unroll
    for (int r = 0; r < 8; r++) {
      Ss[wave][rb + r][col] = s0[r];
      Ss[wave][rb + r][col + 16] = s1[r];
    }
    __syncthreads();

    // online softmax: lanes 0..15 each own one q row
    if (lane < 16) {
      int row = lane;
      int qrow = q0 + wave * 16 + row;
      float m_old = mrow[wave][row];
      float mnew = m_old;
      float sv[32];
#pragma unroll
      for (int j = 0; j < 32; j++) {
        float s = Ss[wave][row][j] * inv_scale;
        if (kt + j > qrow) s = -1e30f;                // causal mask
        sv[j] = s;
        mnew = fmaxf(mnew, s);
      }
      float alpha = __expf(m_old - mnew);
      float l = lrow[wave][row] * alpha;
#pragma unroll
      for (int j = 0; j < 32; j++) {
        float p = __expf(sv[j] - mnew);
        Ps[wave][row][j] = (__bf16)p;
        l += p;
      }
      mrow[wave][row] = mnew;
      lrow[wave][row] = l;
      arow[wave][row] = alpha;
    }
    __syncthreads();

    // rescale O and accumulate P @ V (single K=32 WMMA per 16-col d block)
    float ar[8];
#pragma unroll
    for (int r = 0; r < 8; r++) ar[r] = arow[wave][rb + r];
    v16bf ap = load_frag_a(&Ps[wave][0][0], 40, 0, lane);
#pragma unroll
    for (int t = 0; t < 8; t++) {
#pragma unroll
      for (int r = 0; r < 8; r++) O[t][r] *= ar[r];
      v16bf bv = load_frag_b(&Vst[t * 16][0], 40, lane);
      O[t] = wmma_bf16(ap, bv, O[t]);
    }
  }

  if (lane < 16) arow[wave][lane] = 1.f / (lrow[wave][lane] + 1e-6f);
  __syncthreads();
  float rinv[8];
#pragma unroll
  for (int r = 0; r < 8; r++) rinv[r] = arow[wave][rb + r];
#pragma unroll
  for (int t = 0; t < 8; t++)
#pragma unroll
    for (int r = 0; r < 8; r++) {
      int qrow = q0 + wave * 16 + rb + r;
      Out[((size_t)b * T_ + qrow) * (H_ * DH_) + h * DH_ + t * 16 + col] =
          (__bf16)(O[t][r] * rinv[r]);
    }
}

// ---------------- orchestration ----------------
extern "C" void kernel_launch(void* const* d_in, const int* in_sizes, int n_in,
                              void* d_out, int out_size, void* d_ws, size_t ws_size,
                              hipStream_t stream) {
  (void)in_sizes; (void)n_in; (void)out_size; (void)ws_size;
  const float* x    = (const float*)d_in[0];
  /* d_in[1] = mask: causal, handled analytically */
  const float* ln_w = (const float*)d_in[2];
  const float* wq   = (const float*)d_in[3];
  const float* wk   = (const float*)d_in[4];
  const float* wv   = (const float*)d_in[5];
  const float* wo   = (const float*)d_in[6];
  const float* ffln = (const float*)d_in[7];
  const float* wg   = (const float*)d_in[8];
  const float* w1   = (const float*)d_in[9];
  const float* w2   = (const float*)d_in[10];
  float* out = (float*)d_out;
  char* ws = (char*)d_ws;

  // workspace layout (bytes)
  float*  q_f    = (float*)(ws + 0);                 // 33,554,432
  float*  k_f    = (float*)(ws + 33554432);          // 16,777,216
  float*  v_f    = (float*)(ws + 50331648);          // 16,777,216
  __bf16* attn_b = (__bf16*)(ws + 0);                // reuses q_f region after RoPE
  __bf16* hmul   = (__bf16*)(ws + 0);                // reuses region after WO GEMM
  __bf16* qb     = (__bf16*)(ws + 67108864);         // 16,777,216
  __bf16* kb     = (__bf16*)(ws + 83886080);         //  8,388,608
  __bf16* vb     = (__bf16*)(ws + 92274688);         //  8,388,608
  __bf16* xln    = (__bf16*)(ws + 100663296);        // 16,777,216 (also h)
  float*  res1   = (float*)(ws + 117440512);         // 33,554,432
  __bf16* wb0    = (__bf16*)(ws + 150994944);        // 23,068,672
  __bf16* wb1    = (__bf16*)(ws + 174063616);        // 23,068,672

  // 1) RMSNorm -> bf16
  rmsnorm_kernel<<<M_, 256, 0, stream>>>(x, ln_w, xln, E_);

  // 2) Q/K/V projections (convert weight jit, WMMA GEMM -> fp32)
  conv_kernel<<<(E_ * E_) / 256, 256, 0, stream>>>(wq, wb0, (size_t)E_ * E_);
  gemm_kernel<<<dim3(E_ / 128, M_ / 128), 256, 0, stream>>>(xln, wb0, q_f, nullptr, M_, E_, E_);
  conv_kernel<<<(E_ * 1024) / 256, 256, 0, stream>>>(wk, wb0, (size_t)E_ * 1024);
  gemm_kernel<<<dim3(1024 / 128, M_ / 128), 256, 0, stream>>>(xln, wb0, k_f, nullptr, M_, 1024, E_);
  conv_kernel<<<(E_ * 1024) / 256, 256, 0, stream>>>(wv, wb0, (size_t)E_ * 1024);
  gemm_kernel<<<dim3(1024 / 128, M_ / 128), 256, 0, stream>>>(xln, wb0, v_f, nullptr, M_, 1024, E_);

  // 3) RoPE + head transpose -> bf16 [B,h,T,D]
  rope_kernel<<<(B_ * T_ * H_ * 64) / 256, 256, 0, stream>>>(q_f, qb, H_);
  rope_kernel<<<(B_ * T_ * KVH_ * 64) / 256, 256, 0, stream>>>(k_f, kb, KVH_);
  vtrans_kernel<<<(B_ * T_ * KVH_ * 128) / 256, 256, 0, stream>>>(v_f, vb, KVH_);

  // 4) causal flash attention -> bf16 [B,T,H*D]
  attn_kernel<<<dim3(T_ / 64, H_, B_), 128, 0, stream>>>(qb, kb, vb, attn_b);

  // 5) WO GEMM + residual add -> res1 (fp32)
  conv_kernel<<<(E_ * E_) / 256, 256, 0, stream>>>(wo, wb0, (size_t)E_ * E_);
  gemm_kernel<<<dim3(E_ / 128, M_ / 128), 256, 0, stream>>>(attn_b, wb0, res1, x, M_, E_, E_);

  // 6) RMSNorm(res1) -> h (bf16, reuse xln slot)
  rmsnorm_kernel<<<M_, 256, 0, stream>>>(res1, ffln, xln, E_);

  // 7) fused gate+up GEMM with SiLU*mul -> bf16
  conv_kernel<<<(E_ * HID_) / 256, 256, 0, stream>>>(wg, wb0, (size_t)E_ * HID_);
  conv_kernel<<<(E_ * HID_) / 256, 256, 0, stream>>>(w1, wb1, (size_t)E_ * HID_);
  gateup_kernel<<<dim3(HID_ / 64, M_ / 128), 256, 0, stream>>>(xln, wb0, wb1, hmul, M_, HID_, E_);

  // 8) down GEMM + residual -> out (fp32)
  conv_kernel<<<(HID_ * E_) / 256, 256, 0, stream>>>(w2, wb0, (size_t)HID_ * E_);
  gemm_kernel<<<dim3(E_ / 128, M_ / 128), 256, 0, stream>>>(hmul, wb0, out, res1, M_, E_, HID_);
}
